// LSTMFeatureExtractor_80187039416893
// MI455X (gfx1250) — matrix operand order repaired and checked by reading, not compiled, not run on hardware
//
#include <hip/hip_runtime.h>
#include <hip/hip_bf16.h>

typedef __attribute__((ext_vector_type(16))) _Float16 v16h;
typedef __attribute__((ext_vector_type(8)))  float    v8f;

#define B_TOT 512
#define T_LEN 512
#define HID   128
#define G4    512   // 4*H

// ---------------------------------------------------------------------------
// WMMA wrapper: D = A(16x32 f16) * B(32x16 f16) + C(16x16 f32)
// ---------------------------------------------------------------------------
__device__ __forceinline__ v8f wmma_f16(v16h a, v16h b, v8f c) {
    return __builtin_amdgcn_wmma_f32_16x16x32_f16(
        /*neg_a=*/false, a, /*neg_b=*/false, b,
        /*c_mod=*/(short)0, c, /*reuse_a=*/false, /*reuse_b=*/false);
}

// A-fragment (16x32, f16) from a row-major tile with given row stride (halves).
__device__ __forceinline__ v16h load_afrag(const _Float16* base, int rowStride,
                                           int kbase, int lane) {
    int m  = lane & 15;
    int hi = (lane >> 4) & 1;
    const _Float16* p = base + m * rowStride + kbase + hi * 8;
    v16h f;
#pragma unroll
    for (int e = 0; e < 8; ++e) { f[e] = p[e]; f[e + 8] = p[16 + e]; }
    return f;
}

// B-fragment (32x16, f16) for D = X * W^T (col n of B = row n of W; f32->f16).
__device__ __forceinline__ v16h load_bfrag_w(const float* __restrict__ W,
                                             int nrow, int kbase, int lane) {
    const float* p = W + (size_t)nrow * HID + kbase + ((lane >= 16) ? 8 : 0);
    v16h f;
#pragma unroll
    for (int e = 0; e < 8; ++e) { f[e] = (_Float16)p[e]; f[e + 8] = (_Float16)p[16 + e]; }
    return f;
}

// Branchless activations: v_exp_f32 + v_rcp_f32 only (no libm branch ladders).
__device__ __forceinline__ float sigm(float x) {
    return __builtin_amdgcn_rcpf(1.f + __expf(-x));
}
__device__ __forceinline__ float tanh_fast(float x) {
    return 1.f - 2.f * __builtin_amdgcn_rcpf(__expf(2.f * x) + 1.f);
}

// ---- gfx1250 async copy (global -> LDS), tracked with ASYNCcnt -------------
__device__ __forceinline__ unsigned lds_off(const void* p) {
    // generic LDS address: low 32 bits are the LDS byte offset
    return (unsigned)(unsigned long long)(uintptr_t)p;
}
__device__ __forceinline__ void async_copy_b64(unsigned ldsoff, const void* g) {
    unsigned long long ga = (unsigned long long)(uintptr_t)g;
    asm volatile("global_load_async_to_lds_b64 %0, %1, off"
                 :: "v"(ldsoff), "v"(ga) : "memory");
}
__device__ __forceinline__ void wait_async0() {
    asm volatile("s_wait_asynccnt 0x0" ::: "memory");
}

// ---------------------------------------------------------------------------
// Kernel 1: y[B,128,T] fp32  ->  x0[B,T,128] f16   (LDS tile transpose)
// ---------------------------------------------------------------------------
__global__ __launch_bounds__(256)
void transpose_convert_kernel(const float* __restrict__ y, _Float16* __restrict__ x0) {
    __shared__ float tile[32][33];
    const int b  = blockIdx.x;
    const int t0 = blockIdx.y * 32;
    const int h0 = blockIdx.z * 32;
    const int tx = threadIdx.x;   // 0..31
    const int ty = threadIdx.y;   // 0..7
#pragma unroll
    for (int i = 0; i < 32; i += 8)
        tile[ty + i][tx] = y[((size_t)b * HID + (h0 + ty + i)) * T_LEN + t0 + tx];
    __syncthreads();
#pragma unroll
    for (int i = 0; i < 32; i += 8)
        x0[((size_t)b * T_LEN + (t0 + ty + i)) * HID + h0 + tx] = (_Float16)tile[tx][ty + i];
}

// ---------------------------------------------------------------------------
// Kernel 2: one LSTM layer. Persistent weight fragments in VGPRs; x_t tiles
// double-buffered in LDS and prefetched with async-to-LDS DMA; 2 barriers/step.
// Grid: 32 blocks (16-row batch tiles) x 512 threads (16 waves).
// ---------------------------------------------------------------------------
__global__ __launch_bounds__(512, 1)
void lstm_layer_kernel(const _Float16* __restrict__ xin,   // [B,T,H] f16
                       const float* __restrict__ Wih,      // [4H,H] f32
                       const float* __restrict__ Whh,      // [4H,H] f32
                       const float* __restrict__ bih,      // [4H]
                       const float* __restrict__ bhh,      // [4H]
                       _Float16* __restrict__ hout)        // [B,T,H] f16
{
    __shared__ _Float16 xtile[2][16][HID + 8]; // double buffer, bank-padded
    __shared__ _Float16 htile[16][HID + 8];
    __shared__ float    gates[16][G4 + 16];
    __shared__ float    bias[G4];

    const int tid  = threadIdx.x;
    const int wave = tid >> 5;
    const int lane = tid & 31;
    const int b0   = blockIdx.x * 16;

    // Cell/thread ownership: thread owns (batch row cb, hidden ch..ch+3).
    const int cb = tid >> 5;
    const int ch = (tid & 31) * 4;

    // Prefetch x_0 tile (8 bytes/thread) straight into LDS via async DMA.
    const _Float16* gx = xin + ((size_t)(b0 + cb) * T_LEN) * HID + ch;
    const unsigned xoff0 = lds_off(&xtile[0][cb][ch]);
    const unsigned xoff1 = lds_off(&xtile[1][cb][ch]);
    async_copy_b64(xoff0, gx);

    bias[tid] = bih[tid] + bhh[tid];
    for (int i = tid; i < 16 * (HID + 8); i += 512)
        (&htile[0][0])[i] = (_Float16)0.f;

    // Persistent weight B-fragments: 2 col tiles x 4 k-blocks x {Wih,Whh}.
    v16h wih_f[2][4], whh_f[2][4];
#pragma unroll
    for (int tile = 0; tile < 2; ++tile) {
        const int nrow = 32 * wave + tile * 16 + (lane & 15);
#pragma unroll
        for (int kb = 0; kb < 4; ++kb) {
            wih_f[tile][kb] = load_bfrag_w(Wih, nrow, kb * 32, lane);
            whh_f[tile][kb] = load_bfrag_w(Whh, nrow, kb * 32, lane);
        }
    }

    float c4[4] = {0.f, 0.f, 0.f, 0.f};

    for (int t = 0; t < T_LEN; ++t) {
        const int buf = t & 1;
        wait_async0();        // my slice of x_t has landed
        __syncthreads();      // everyone's slice visible; htile(t-1) published

        // Prefetch x_{t+1} into the other buffer while we compute.
        if (t + 1 < T_LEN)
            async_copy_b64((t + 1) & 1 ? xoff1 : xoff0,
                           gx + (size_t)(t + 1) * HID);

        // ---- gates = x_t*Wih^T + h*Whh^T : 16 WMMAs per wave ----
        const _Float16* xb = &xtile[buf][0][0];
#pragma unroll
        for (int tile = 0; tile < 2; ++tile) {
            v8f acc = {};
#pragma unroll
            for (int kb = 0; kb < 4; ++kb) {
                v16h ax = load_afrag(xb, HID + 8, kb * 32, lane);
                acc = wmma_f16(ax, wih_f[tile][kb], acc);
                v16h ah = load_afrag(&htile[0][0], HID + 8, kb * 32, lane);
                acc = wmma_f16(ah, whh_f[tile][kb], acc);
            }
            const int col   = 32 * wave + tile * 16 + (lane & 15);
            const int mbase = (lane >> 4) ? 8 : 0;
#pragma unroll
            for (int v = 0; v < 8; ++v) gates[mbase + v][col] = acc[v];
        }
        __syncthreads();

        // ---- elementwise LSTM cell update (4 cells per thread) ----
        _Float16* hdst = &htile[cb][ch];
        _Float16* gdst = hout + ((size_t)(b0 + cb) * T_LEN + t) * HID + ch;
#pragma unroll
        for (int j = 0; j < 4; ++j) {
            const int hid = ch + j;
            float ig = sigm(gates[cb][hid]              + bias[hid]);
            float fg = sigm(gates[cb][HID   + hid]      + bias[HID + hid]);
            float gg = tanh_fast(gates[cb][2*HID + hid] + bias[2*HID + hid]);
            float og = sigm(gates[cb][3*HID + hid]      + bias[3*HID + hid]);
            c4[j] = fg * c4[j] + ig * gg;
            float hv = og * tanh_fast(c4[j]);
            hdst[j] = (_Float16)hv;
            gdst[j] = (_Float16)hv;
        }
        // no third barrier: next iteration's sync-A orders htile/gates reuse
    }
}

// ---------------------------------------------------------------------------
// Kernel 3: a = h1 @ Wa^T + ba ; out[b,:] = sum_t a*relu(a).
// Grid: 512 blocks (one per b) x 256 threads (8 waves).
// ---------------------------------------------------------------------------
__global__ __launch_bounds__(256, 1)
void proj_reduce_kernel(const _Float16* __restrict__ h1,  // [B,T,H] f16
                        const float* __restrict__ Wa,     // [H,H]
                        const float* __restrict__ ba,     // [H]
                        float* __restrict__ out)          // [B,H]
{
    const int tid  = threadIdx.x;
    const int wave = tid >> 5;
    const int lane = tid & 31;
    const int b    = blockIdx.x;
    const int ncol = wave * 16 + (lane & 15);

    v16h wf[4];
#pragma unroll
    for (int kb = 0; kb < 4; ++kb) wf[kb] = load_bfrag_w(Wa, ncol, kb * 32, lane);
    const float bv = ba[ncol];

    float s = 0.f;
    for (int t0 = 0; t0 < T_LEN; t0 += 16) {
        const _Float16* abase = h1 + ((size_t)b * T_LEN + t0) * HID;
        v8f acc = {};
#pragma unroll
        for (int kb = 0; kb < 4; ++kb) {
            v16h af = load_afrag(abase, HID, kb * 32, lane);
            acc = wmma_f16(af, wf[kb], acc);
        }
#pragma unroll
        for (int v = 0; v < 8; ++v) {
            float a = acc[v] + bv;
            s += (a > 0.f) ? a * a : 0.f;   // a * relu(a)
        }
    }
    s += __shfl_xor(s, 16, 32);
    if (lane < 16) out[(size_t)b * HID + ncol] = s;
}

// ---------------------------------------------------------------------------
extern "C" void kernel_launch(void* const* d_in, const int* in_sizes, int n_in,
                              void* d_out, int out_size, void* d_ws, size_t ws_size,
                              hipStream_t stream) {
    const float* y    = (const float*)d_in[0];
    const float* Wih0 = (const float*)d_in[1];
    const float* Whh0 = (const float*)d_in[2];
    const float* bih0 = (const float*)d_in[3];
    const float* bhh0 = (const float*)d_in[4];
    const float* Wih1 = (const float*)d_in[5];
    const float* Whh1 = (const float*)d_in[6];
    const float* bih1 = (const float*)d_in[7];
    const float* bhh1 = (const float*)d_in[8];
    const float* Wa   = (const float*)d_in[9];
    const float* ba   = (const float*)d_in[10];

    const size_t seqBytes = (size_t)B_TOT * T_LEN * HID * sizeof(_Float16); // 64 MB
    _Float16* x0 = (_Float16*)d_ws;                       // layer-0 input
    _Float16* h0 = (_Float16*)((char*)d_ws + seqBytes);   // layer-0 output
    _Float16* h1 = x0;                                    // layer-1 output reuses x0

    transpose_convert_kernel<<<dim3(B_TOT, T_LEN / 32, HID / 32), dim3(32, 8), 0, stream>>>(y, x0);
    lstm_layer_kernel<<<B_TOT / 16, 512, 0, stream>>>(x0, Wih0, Whh0, bih0, bhh0, h0);
    lstm_layer_kernel<<<B_TOT / 16, 512, 0, stream>>>(h0, Wih1, Whh1, bih1, bhh1, h1);
    proj_reduce_kernel<<<B_TOT, 256, 0, stream>>>(h1, Wa, ba, (float*)d_out);
}
